// ManyBodyVoxel_71631464563029
// MI455X (gfx1250) — matrix operand order repaired and checked by reading, not compiled, not run on hardware
//
#include <hip/hip_runtime.h>
#include <hip/hip_bf16.h>

typedef __attribute__((ext_vector_type(16))) _Float16 v16h;
typedef __attribute__((ext_vector_type(8)))  _Float16 v8h;
typedef __attribute__((ext_vector_type(8)))  float    v8f;

#define NB   4    // batch
#define NA   64   // atoms
#define NN   32   // neighbors
#define NG   16   // grid points per axis

// One workgroup (256 threads = 8 wave32) per (b,a) pair.
// Separable-Gaussian + WMMA formulation:
//   out[c][(ix,iy),iz] = sum_n exy[n,(ix,iy)] * (ez[n,iz]*w[n,c])
// -> 16x16x32 f16 WMMA tiles, f32 accumulate.
__global__ __launch_bounds__(256) void mbv_wmma_kernel(
    const float* __restrict__ dvec,   // (B,A,N,3)
    const float* __restrict__ dist,   // (B,A,N)
    const int*   __restrict__ zatom,  // (B,A)
    const int*   __restrict__ neigh,  // (B,A,N)
    const int*   __restrict__ nmask,  // (B,A,N)
    float*       __restrict__ out)    // (B,A,4,16,16,16)
{
    __shared__ float    sEx [2][3][NN][NG];   // [length][dim][n][i]  12 KB
    __shared__ float    sW  [2][2][NN];       // [length][type][n]    0.5 KB
    __shared__ _Float16 exyT[2][256][NN];     // A tiles: rows=(ix,iy), cols=n  32 KB
    __shared__ _Float16 ezwT[2][2][NG][NN];   // B tiles (transposed): [iz][n]  4 KB

    const int ba = blockIdx.x;        // b*NA + a
    const int b  = ba >> 6;
    const int t  = threadIdx.x;

    // ---------------- Phase 1: 1-D Gaussians + cutoff weights ----------------
    if (t < 192) {
        const int d = t >> 6;         // dim 0..2
        const int l = (t >> 5) & 1;   // grid length 0..1
        const int n = t & 31;         // neighbor
        // All per-length constants as literal selects (no rodata load, no fdiv):
        // spacing = L/15, coeff = -0.5/spacing^2 = -112.5/L^2, o0 = -L/2
        const float spacing = l ? (10.0f / 15.0f) : (6.0f / 15.0f);
        const float coeff   = l ? -1.125f : -3.125f;
        const float o0      = l ? -5.0f : -3.0f;
        const float x       = dvec[(ba * NN + n) * 3 + d];
        #pragma unroll
        for (int i = 0; i < NG; ++i) {
            float g  = o0 + spacing * (float)i;
            float dd = g - x;
            sEx[l][d][n][i] = __expf(coeff * dd * dd);   // v_exp_f32
        }
        if (d == 0) {
            const float r       = dist[ba * NN + n];
            const float cutoff  = l ? 5.0f : 3.0f;       // L/2
            const float piOverC = l ? 0.62831853071795864f   // pi/5
                                    : 1.04719755119659774f;  // pi/3
            float C = 0.5f * (__cosf(r * piOverC) + 1.0f);
            C = (r < cutoff) ? C : 0.0f;
            const int zi = neigh[ba * NN + n];
            const int zn = zatom[b * NA + zi] * nmask[ba * NN + n];
            sW[l][0][n] = (zn == 1) ? C : 0.0f;
            sW[l][1][n] = (zn == 8) ? C : 0.0f;
        }
    }
    __syncthreads();

    // ---------------- Phase 2: stage f16 WMMA operand tiles in LDS ----------
    {
        const int ix = t >> 4;
        const int iy = t & 15;
        #pragma unroll
        for (int l = 0; l < 2; ++l) {
            #pragma unroll
            for (int n = 0; n < NN; ++n) {
                exyT[l][t][n] = (_Float16)(sEx[l][0][n][ix] * sEx[l][1][n][iy]);
            }
        }
        // ezwT: 2*2*16*32 = 2048 entries, 8 per thread
        #pragma unroll
        for (int k = 0; k < 8; ++k) {
            int e  = t * 8 + k;
            int n  = e & 31;
            int iz = (e >> 5) & 15;
            int ty = (e >> 9) & 1;
            int l  = (e >> 10) & 1;
            ezwT[l][ty][iz][n] = (_Float16)(sEx[l][2][n][iz] * sW[l][ty][n]);
        }
    }
    __syncthreads();

    // ---------------- Phase 3: WMMA batch GEMM ------------------------------
    const int wave    = t >> 5;
    const int lane    = t & 31;
    const int hi      = lane >> 4;        // lane half: 0 => lanes 0-15
    const int lane16  = lane & 15;

    const int l       = wave >> 2;        // grid length
    const int ty      = (wave >> 1) & 1;  // atom type
    const int grpBase = (wave & 1) * 8;   // which 8 row-groups of 16

    // B fragment (32x16, K=n, N=iz): lane holds column N=lane16,
    // K = hi*8 .. hi*8+7 and 16+hi*8 .. 16+hi*8+7 in ezwT[...][lane16][...]
    v16h bfrag;
    {
        const v8h* bpv = (const v8h*)(&ezwT[l][ty][lane16][0]);  // 16B-aligned rows
        v8h b0 = bpv[hi];
        v8h b1 = bpv[2 + hi];
        bfrag = __builtin_shufflevector(b0, b1, 0, 1, 2, 3, 4, 5, 6, 7,
                                                8, 9, 10, 11, 12, 13, 14, 15);
    }

    float* outc = out + ((size_t)ba * 4 + (size_t)(l * 2 + ty)) * 4096;

    #pragma unroll
    for (int g = 0; g < 8; ++g) {
        const int grp = grpBase + g;
        const int row = grp * 16 + lane16;          // M index = (ix,iy) flat
        // A fragment (16x32): lanes0-15 VGPR0-3 K=0-7, VGPR4-7 K=16-23;
        //                     lanes16-31 shift K by +8.
        const v8h* apv = (const v8h*)(&exyT[l][row][0]);  // 64B row stride
        v8h a0 = apv[hi];          // K = hi*8 .. hi*8+7
        v8h a1 = apv[2 + hi];      // K = 16+hi*8 .. 16+hi*8+7
        v16h afrag = __builtin_shufflevector(a0, a1, 0, 1, 2, 3, 4, 5, 6, 7,
                                                     8, 9, 10, 11, 12, 13, 14, 15);

        v8f acc = {};
        acc = __builtin_amdgcn_wmma_f32_16x16x32_f16(
            /*neg_a=*/false, afrag, /*neg_b=*/false, bfrag,
            /*c_mod=*/(short)0, acc, /*reuse_a=*/false, /*reuse_b=*/false);

        // D layout: VGPR v -> M = v + 8*hi, N = lane16
        #pragma unroll
        for (int v = 0; v < 8; ++v) {
            const int m = v + hi * 8;
            outc[(size_t)((grp * 16 + m) * 16 + lane16)] = acc[v];
        }
    }
}

extern "C" void kernel_launch(void* const* d_in, const int* in_sizes, int n_in,
                              void* d_out, int out_size, void* d_ws, size_t ws_size,
                              hipStream_t stream) {
    const float* dvec  = (const float*)d_in[0];  // distance_vector (B,A,N,3)
    const float* dist  = (const float*)d_in[1];  // distances       (B,A,N)
    const int*   zatom = (const int*)d_in[2];    // atomic_numbers  (B,A)
    const int*   neigh = (const int*)d_in[3];    // neighbors       (B,A,N)
    const int*   nmask = (const int*)d_in[4];    // neighbor_mask   (B,A,N)
    float* out = (float*)d_out;                  // (B,A,4,16,16,16)

    mbv_wmma_kernel<<<dim3(NB * NA), dim3(256), 0, stream>>>(
        dvec, dist, zatom, neigh, nmask, out);
}
